// C2BM_30537217474758
// MI455X (gfx1250) — matrix operand order repaired and checked by reading, not compiled, not run on hardware
//
#include <hip/hip_runtime.h>

// ---------------------------------------------------------------------------
// Types for CDNA5 WMMA (wave32): v_wmma_f32_16x16x32_bf16
// ---------------------------------------------------------------------------
typedef __attribute__((ext_vector_type(16))) __bf16 v16bf;
typedef __attribute__((ext_vector_type(8)))  float  v8f;

union Frag32B { uint4 q[2]; v16bf v; };

// ---------------------------------------------------------------------------
// Problem constants
// ---------------------------------------------------------------------------
#define BATCH   8192
#define D_IN    2048
#define D_H     1024

// GEMM tiling
#define BM   128
#define BN   128
#define BK   32
#define LDT  40        // padded LDS stride in bf16 elems; 80B = 16*5 -> b128 ok
#define TILE_ELEMS ((BM + BN) * LDT)

// LDS byte offset of a generic pointer into __shared__ memory:
// flat LDS address = {SHARED_BASE, offset[31:0]} per ISA aperture rules.
__device__ __forceinline__ unsigned lds_off(const void* p) {
    return (unsigned)(uintptr_t)p;
}

__device__ __forceinline__ void async_copy_b128(const void* gsrc, void* ldst) {
    asm volatile("global_load_async_to_lds_b128 %0, %1, off"
                 :: "v"(lds_off(ldst)),
                    "v"((unsigned long long)(uintptr_t)gsrc)
                 : "memory");
}

__device__ __forceinline__ void wait_async0() {
    asm volatile("s_wait_asynccnt 0x0" ::: "memory");
}

// ---------------------------------------------------------------------------
// Prep kernels: make bf16 K-major ("B transposed") weight copies.
// ---------------------------------------------------------------------------
// W_enc [2048][1024] f32  ->  WencT [1024][2048] bf16  (WencT[n][k] = W[k][n])
__global__ __launch_bounds__(256) void prep_wenc(const float* __restrict__ W,
                                                 __bf16* __restrict__ Wt) {
    int idx = blockIdx.x * 256 + threadIdx.x;     // over 1024*2048
    int n = idx >> 11;
    int k = idx & 2047;
    Wt[idx] = (__bf16)W[(size_t)k * D_H + n];
}

// Wv [8][1024][256] f32 -> WvT [2048][1024] bf16 (WvT[r*256+o][i] = Wv[r][i][o])
__global__ __launch_bounds__(256) void prep_wv(const float* __restrict__ Wv,
                                               __bf16* __restrict__ Wt) {
    int idx = blockIdx.x * 256 + threadIdx.x;     // over 2048*1024
    int col = idx >> 10;                           // r*256+o
    int i   = idx & 1023;
    int r = col >> 8, o = col & 255;
    Wt[idx] = (__bf16)Wv[((size_t)r * D_H + i) * 256 + o];
}

// ---------------------------------------------------------------------------
// WMMA GEMM: out[M][N] = leaky(A[M][K] * Bt[N][K]^T + bias[N]) in bf16.
// AKIND: 0 = A is f32 row-major (VALU cvt, reg-staged), 1 = A is bf16 (async).
// 256 threads = 8 waves; waves in 4(M) x 2(N); each wave owns 2x4 16x16 tiles.
// Double-buffered LDS; B tiles (and bf16 A tiles) via async global->LDS DMA.
// ---------------------------------------------------------------------------
template <int AKIND>
__global__ __launch_bounds__(256) void gemm_bias_leaky(
    const void* __restrict__ Aptr, const __bf16* __restrict__ Bt,
    const float* __restrict__ bias, __bf16* __restrict__ out,
    int M, int N, int K)
{
    (void)M;
    __shared__ __align__(16) __bf16 smem[2 * TILE_ELEMS];

    const int tid  = threadIdx.x;
    const int lane = tid & 31;
    const int wid  = tid >> 5;
    const int bm = blockIdx.x * BM;
    const int bn = blockIdx.y * BN;
    const int wm = (wid >> 1) * 32;    // wave M offset in block tile
    const int wn = (wid & 1) * 64;     // wave N offset in block tile

    const int mrow = lane & 15;        // fragment row / B column
    const int kh   = lane >> 4;        // K-half select

    // 128x32 bf16 tile = 512 chunks of 8 elems (16B); 256 threads -> 2 iters.
    // chunk j: row = j>>2, c8 = (j&3)*8
    const __bf16* Abf = (const __bf16*)Aptr;
    const float*  Af  = (const float*)Aptr;

    // ---- prologue: stage tile 0 into buffer 0 -----------------------------
    {
        __bf16* As = smem;
        __bf16* Bs = smem + BM * LDT;
        #pragma unroll
        for (int it = 0; it < 2; ++it) {
            int j = tid + it * 256;
            int row = j >> 2, c8 = (j & 3) * 8;
            async_copy_b128(&Bt[(size_t)(bn + row) * K + c8], &Bs[row * LDT + c8]);
        }
        if constexpr (AKIND == 0) {
            #pragma unroll
            for (int it = 0; it < 4; ++it) {       // float4 chunks (4 elems)
                int j = tid + it * 256;
                int row = j >> 3, c4 = (j & 7) * 4;
                float4 v = *(const float4*)&Af[(size_t)(bm + row) * K + c4];
                __bf16* dst = &As[row * LDT + c4];
                dst[0] = (__bf16)v.x; dst[1] = (__bf16)v.y;
                dst[2] = (__bf16)v.z; dst[3] = (__bf16)v.w;
            }
        } else {
            #pragma unroll
            for (int it = 0; it < 2; ++it) {
                int j = tid + it * 256;
                int row = j >> 2, c8 = (j & 3) * 8;
                async_copy_b128(&Abf[(size_t)(bm + row) * K + c8], &As[row * LDT + c8]);
            }
        }
    }

    v8f acc[2][4] = {};
    int buf = 0;

    for (int k0 = 0; k0 < K; k0 += BK) {
        wait_async0();
        __syncthreads();

        const bool has_next = (k0 + BK) < K;
        __bf16* Asc = smem + buf * TILE_ELEMS;
        __bf16* Bsc = Asc + BM * LDT;
        __bf16* Asn = smem + (buf ^ 1) * TILE_ELEMS;
        __bf16* Bsn = Asn + BM * LDT;
        const int kn = k0 + BK;

        // ---- issue next-tile loads first (hide behind WMMA) ---------------
        float4 areg[4];
        if (has_next) {
            #pragma unroll
            for (int it = 0; it < 2; ++it) {
                int j = tid + it * 256;
                int row = j >> 2, c8 = (j & 3) * 8;
                async_copy_b128(&Bt[(size_t)(bn + row) * K + kn + c8],
                                &Bsn[row * LDT + c8]);
            }
            if constexpr (AKIND == 0) {
                #pragma unroll
                for (int it = 0; it < 4; ++it) {
                    int j = tid + it * 256;
                    int row = j >> 3, c4 = (j & 7) * 4;
                    areg[it] = *(const float4*)&Af[(size_t)(bm + row) * K + kn + c4];
                }
            } else {
                #pragma unroll
                for (int it = 0; it < 2; ++it) {
                    int j = tid + it * 256;
                    int row = j >> 2, c8 = (j & 3) * 8;
                    async_copy_b128(&Abf[(size_t)(bm + row) * K + kn + c8],
                                    &Asn[row * LDT + c8]);
                }
            }
        }

        // ---- build A fragments (ISA 16-bit A 16x32 layout) ----------------
        // lane<16: elems = K[kh*8 .. +7], K[kh*8+16 .. +23]
        v16bf afrag[2];
        #pragma unroll
        for (int mi = 0; mi < 2; ++mi) {
            const __bf16* ar = &Asc[(wm + mi * 16 + mrow) * LDT + kh * 8];
            Frag32B u;
            u.q[0] = *(const uint4*)(ar);
            u.q[1] = *(const uint4*)(ar + 16);
            afrag[mi] = u.v;
        }
        // ---- B fragments (col = lane&15, K = kh*8+0..7, kh*8+16..23) + WMMA
        #pragma unroll
        for (int ni = 0; ni < 4; ++ni) {
            const __bf16* br = &Bsc[(wn + ni * 16 + mrow) * LDT + kh * 8];
            Frag32B u;
            u.q[0] = *(const uint4*)(br);
            u.q[1] = *(const uint4*)(br + 16);
            v16bf bfrag = u.v;
            #pragma unroll
            for (int mi = 0; mi < 2; ++mi) {
                acc[mi][ni] = __builtin_amdgcn_wmma_f32_16x16x32_bf16(
                    false, afrag[mi], false, bfrag, (short)0, acc[mi][ni],
                    false, false);
            }
        }

        // ---- commit reg-staged next A tile (f32 path) ---------------------
        if (has_next) {
            if constexpr (AKIND == 0) {
                #pragma unroll
                for (int it = 0; it < 4; ++it) {
                    int j = tid + it * 256;
                    int row = j >> 3, c4 = (j & 7) * 4;
                    __bf16* dst = &Asn[row * LDT + c4];
                    dst[0] = (__bf16)areg[it].x; dst[1] = (__bf16)areg[it].y;
                    dst[2] = (__bf16)areg[it].z; dst[3] = (__bf16)areg[it].w;
                }
            }
        }
        buf ^= 1;
    }

    // ---- epilogue: bias + leaky_relu -> bf16 (C layout: row=v+8*kh) -------
    #pragma unroll
    for (int mi = 0; mi < 2; ++mi) {
        #pragma unroll
        for (int ni = 0; ni < 4; ++ni) {
            int col  = bn + wn + ni * 16 + mrow;
            int row0 = bm + wm + mi * 16 + kh * 8;
            float bc = bias[col];
            #pragma unroll
            for (int v = 0; v < 8; ++v) {
                float x = acc[mi][ni][v] + bc;
                x = x > 0.0f ? x : 0.01f * x;
                out[(size_t)(row0 + v) * N + col] = (__bf16)x;
            }
        }
    }
}

// ---------------------------------------------------------------------------
// Root head: logits = vals[b, r*256 : +256] . Ws[r] + bs[r]; softmax;
// intervention; write p_root to d_out slots [0..7] and pr_flat workspace.
// block = 256 threads = 32 rows x 8 roots.
// ---------------------------------------------------------------------------
__global__ __launch_bounds__(256) void root_head(
    const __bf16* __restrict__ vals, const float* __restrict__ Ws,
    const float* __restrict__ bs, const int* __restrict__ cc,
    const int* __restrict__ mask, float* __restrict__ out,
    float* __restrict__ pr_flat)
{
    int tid = threadIdx.x;
    int r   = tid & 7;
    int b   = blockIdx.x * 32 + (tid >> 3);

    const __bf16* vrow = vals + (size_t)b * 2048 + r * 256;
    const float*  ws   = Ws + r * 1024;            // [256][4]
    float lg[4] = { bs[r * 4 + 0], bs[r * 4 + 1], bs[r * 4 + 2], bs[r * 4 + 3] };
    for (int o = 0; o < 256; ++o) {
        float v = (float)vrow[o];
        lg[0] += v * ws[o * 4 + 0];
        lg[1] += v * ws[o * 4 + 1];
        lg[2] += v * ws[o * 4 + 2];
        lg[3] += v * ws[o * 4 + 3];
    }
    float mx = fmaxf(fmaxf(lg[0], lg[1]), fmaxf(lg[2], lg[3]));
    float e0 = __expf(lg[0] - mx), e1 = __expf(lg[1] - mx);
    float e2 = __expf(lg[2] - mx), e3 = __expf(lg[3] - mx);
    float inv = 1.0f / (e0 + e1 + e2 + e3);
    float p[4] = { e0 * inv, e1 * inv, e2 * inv, e3 * inv };

    int lbl = cc[(size_t)b * 17 + r];
    int m   = mask[(size_t)b * 17 + r];
    #pragma unroll
    for (int i = 0; i < 4; ++i) {
        float pi = (m > 0) ? ((i == lbl) ? 1.0f : 0.0f) : p[i];
        out[((size_t)b * 17 + r) * 4 + i] = pi;
        pr_flat[(size_t)b * 32 + r * 4 + i] = pi;
    }
}

// ---------------------------------------------------------------------------
// Mid-level + task head: one thread per batch row; weights are wave-uniform.
// ---------------------------------------------------------------------------
__global__ __launch_bounds__(256) void mid_task(
    const float* __restrict__ pr_flat,
    const float* __restrict__ W1c, const float* __restrict__ b1c,
    const float* __restrict__ W2c, const float* __restrict__ b2c,
    const float* __restrict__ W1y, const float* __restrict__ b1y,
    const float* __restrict__ W2y, const float* __restrict__ b2y,
    const int* __restrict__ cc, const int* __restrict__ mask,
    float* __restrict__ out)
{
    int b = blockIdx.x * 256 + threadIdx.x;

    float pr[32];
    #pragma unroll
    for (int p = 0; p < 32; ++p) pr[p] = pr_flat[(size_t)b * 32 + p];

    float pm[32];
    for (int n = 0; n < 8; ++n) {
        const float* w1 = W1c + n * (32 * 64);
        const float* w2 = W2c + n * (64 * 4);
        const float* bb = b1c + n * 64;
        float lg[4] = { b2c[n * 4 + 0], b2c[n * 4 + 1],
                        b2c[n * 4 + 2], b2c[n * 4 + 3] };
        for (int j = 0; j < 64; ++j) {
            float h = bb[j];
            #pragma unroll
            for (int p = 0; p < 32; ++p) h += pr[p] * w1[p * 64 + j];
            h = h > 0.0f ? h : 0.01f * h;
            lg[0] += h * w2[j * 4 + 0];
            lg[1] += h * w2[j * 4 + 1];
            lg[2] += h * w2[j * 4 + 2];
            lg[3] += h * w2[j * 4 + 3];
        }
        float mx = fmaxf(fmaxf(lg[0], lg[1]), fmaxf(lg[2], lg[3]));
        float e0 = __expf(lg[0] - mx), e1 = __expf(lg[1] - mx);
        float e2 = __expf(lg[2] - mx), e3 = __expf(lg[3] - mx);
        float inv = 1.0f / (e0 + e1 + e2 + e3);
        float q[4] = { e0 * inv, e1 * inv, e2 * inv, e3 * inv };

        int lbl = cc[(size_t)b * 17 + 8 + n];
        int m   = mask[(size_t)b * 17 + 8 + n];
        #pragma unroll
        for (int i = 0; i < 4; ++i) {
            float qi = (m > 0) ? ((i == lbl) ? 1.0f : 0.0f) : q[i];
            out[((size_t)b * 17 + 8 + n) * 4 + i] = qi;
            pm[n * 4 + i] = qi;
        }
    }

    // task head (no intervention)
    float lg[4] = { b2y[0], b2y[1], b2y[2], b2y[3] };
    for (int j = 0; j < 64; ++j) {
        float h = b1y[j];
        #pragma unroll
        for (int p = 0; p < 32; ++p) h += pm[p] * W1y[p * 64 + j];
        h = h > 0.0f ? h : 0.01f * h;
        lg[0] += h * W2y[j * 4 + 0];
        lg[1] += h * W2y[j * 4 + 1];
        lg[2] += h * W2y[j * 4 + 2];
        lg[3] += h * W2y[j * 4 + 3];
    }
    float mx = fmaxf(fmaxf(lg[0], lg[1]), fmaxf(lg[2], lg[3]));
    float e0 = __expf(lg[0] - mx), e1 = __expf(lg[1] - mx);
    float e2 = __expf(lg[2] - mx), e3 = __expf(lg[3] - mx);
    float inv = 1.0f / (e0 + e1 + e2 + e3);
    #pragma unroll
    for (int i = 0; i < 4; ++i) {
        float e = (i == 0 ? e0 : i == 1 ? e1 : i == 2 ? e2 : e3);
        out[((size_t)b * 17 + 16) * 4 + i] = e * inv;
    }
}

// ---------------------------------------------------------------------------
// Host launcher
// ---------------------------------------------------------------------------
extern "C" void kernel_launch(void* const* d_in, const int* in_sizes, int n_in,
                              void* d_out, int out_size, void* d_ws,
                              size_t ws_size, hipStream_t stream)
{
    (void)in_sizes; (void)n_in; (void)out_size; (void)ws_size;

    const float* x      = (const float*)d_in[0];
    const int*   c      = (const int*)d_in[1];
    const int*   iidx   = (const int*)d_in[2];
    const float* W_enc  = (const float*)d_in[3];
    const float* b_enc  = (const float*)d_in[4];
    const float* Wv     = (const float*)d_in[5];
    const float* bv     = (const float*)d_in[6];
    const float* Ws     = (const float*)d_in[7];
    const float* bs     = (const float*)d_in[8];
    const float* W1c    = (const float*)d_in[9];
    const float* b1c    = (const float*)d_in[10];
    const float* W2c    = (const float*)d_in[11];
    const float* b2c    = (const float*)d_in[12];
    const float* W1y    = (const float*)d_in[13];
    const float* b1y    = (const float*)d_in[14];
    const float* W2y    = (const float*)d_in[15];
    const float* b2y    = (const float*)d_in[16];
    float* out = (float*)d_out;

    // workspace layout (bytes)
    char* ws = (char*)d_ws;
    __bf16* WencT = (__bf16*)(ws + 0);                        //  4 MiB
    __bf16* WvT   = (__bf16*)(ws + (size_t)4  * 1024 * 1024); //  4 MiB
    __bf16* xenc  = (__bf16*)(ws + (size_t)8  * 1024 * 1024); // 16 MiB
    __bf16* vals  = (__bf16*)(ws + (size_t)24 * 1024 * 1024); // 32 MiB
    float*  prfl  = (float*) (ws + (size_t)56 * 1024 * 1024); //  1 MiB

    // 1) weight prep (f32 -> bf16 K-major)
    prep_wenc<<<(D_H * D_IN) / 256, 256, 0, stream>>>(W_enc, WencT);
    prep_wv  <<<(2048 * D_H) / 256, 256, 0, stream>>>(Wv, WvT);

    // 2) encoder GEMM: x[8192,2048] * WencT^T -> xenc bf16 [8192,1024]
    gemm_bias_leaky<0><<<dim3(BATCH / BM, D_H / BN), 256, 0, stream>>>(
        (const void*)x, WencT, b_enc, xenc, BATCH, D_H, D_IN);

    // 3) value-embedding GEMM: xenc[8192,1024] * WvT^T -> vals bf16 [8192,2048]
    gemm_bias_leaky<1><<<dim3(BATCH / BM, 2048 / BN), 256, 0, stream>>>(
        (const void*)xenc, WvT, bv, vals, BATCH, 2048, D_H);

    // 4) root scorer + softmax + intervention
    root_head<<<BATCH / 32, 256, 0, stream>>>(vals, Ws, bs, c, iidx, out, prfl);

    // 5) mid concepts + task head
    mid_task<<<BATCH / 256, 256, 0, stream>>>(prfl, W1c, b1c, W2c, b2c,
                                              W1y, b1y, W2y, b2y, c, iidx, out);
}
